// NodeProcessor_2070174236986
// MI455X (gfx1250) — matrix operand order repaired and checked by reading, not compiled, not run on hardware
//
#include <hip/hip_runtime.h>
#include <hip/hip_bf16.h>
#include <stdint.h>

#define N_NODES 50000
#define N_EDGES 600000
#define D 128          // D_NODE == D_EDGE == HIDDEN == 128
#define LN_EPS 1e-5f

typedef __attribute__((ext_vector_type(16))) __bf16          v16bf;
typedef __attribute__((ext_vector_type(8)))  float           v8f;
typedef __attribute__((ext_vector_type(16))) unsigned short  v16u;
typedef __attribute__((ext_vector_type(8)))  unsigned short  v8u;

// f32 -> bf16: let LLVM's fptrunc lowering pick the native conversion
// instruction if gfx1250 has one (else it emits its own RNE sequence).
__device__ __forceinline__ unsigned short f2bf(float f) {
  __bf16 b = (__bf16)f;
  return __builtin_bit_cast(unsigned short, b);
}

__device__ __forceinline__ v8f wmma_bf16(v16u a, v16u b, v8f c) {
  return __builtin_amdgcn_wmma_f32_16x16x32_bf16(
      false, __builtin_bit_cast(v16bf, a),
      false, __builtin_bit_cast(v16bf, b),
      (short)0, c, false, false);
}

// ---------------------------------------------------------------- zero agg
__global__ void zero_f4(float4* __restrict__ p, int n4) {
  int i = blockIdx.x * blockDim.x + threadIdx.x;
  int stride = gridDim.x * blockDim.x;
  for (; i < n4; i += stride) p[i] = make_float4(0.f, 0.f, 0.f, 0.f);
}

// ------------------------------------------- pack weights into B-frag order
// B-fragment (32x16 bf16, wave32): lane half hi holds K=hi*8+0..7 (elems 0..7)
// and K=hi*8+16..23 (elems 8..15).  Fragment storage: 32 lanes * 16 bf16, so each
// lane reads its 32 bytes contiguously.
__global__ void pack_w(const float* __restrict__ W, unsigned short* __restrict__ out,
                       int Ktot) {
  int idx = blockIdx.x * blockDim.x + threadIdx.x;
  if (idx >= Ktot * D) return;
  int k = idx >> 7;            // row (K)
  int n = idx & 127;           // col (N)
  int kc = k >> 5, kk = k & 31;
  int j  = n >> 4, nn = n & 15;
  int hi = (kk >> 3) & 1, e = (kk & 7) + ((kk >> 4) << 3);
  int frag = kc * 8 + j;
  out[frag * 512 + (nn + hi * 16) * 16 + e] = f2bf(W[idx]);
}

// ---------------------------------------------------------------- scatter
__global__ void scatter_edges(const float* __restrict__ ea, const int* __restrict__ ei,
                              float* __restrict__ agg) {
  int t = blockIdx.x * blockDim.x + threadIdx.x;   // one thread per (edge, 4-col chunk)
  int e = t >> 5;
  int lane = t & 31;
  if (e >= N_EDGES) return;
  int dst = ei[N_EDGES + e];                       // edge_index[1][e]
  const float4 v = *(const float4*)(ea + (size_t)e * D + lane * 4);
  float* a = agg + (size_t)dst * D + lane * 4;
  atomicAdd(a + 0, v.x);
  atomicAdd(a + 1, v.y);
  atomicAdd(a + 2, v.z);
  atomicAdd(a + 3, v.w);
}

// ------------------------------------------------------- fused MLP + LayerNorm
// 128 threads = 4 waves; each wave owns a 16-row node tile.
// A-fragment (16x32 bf16): lane m=lane&15, hi=lane>>4:
//   elems 0..7  -> K = hi*8 + 0..7
//   elems 8..15 -> K = hi*8 + 16..23
__global__ __launch_bounds__(128)
void mgn_mlp(const float* __restrict__ x, const float* __restrict__ agg,
             const unsigned short* __restrict__ w1f,
             const unsigned short* __restrict__ w2f,
             const unsigned short* __restrict__ w3f,
             const float* __restrict__ b1, const float* __restrict__ b2,
             const float* __restrict__ b3, const float* __restrict__ gamma,
             const float* __restrict__ beta, float* __restrict__ out) {
  __shared__ float stageF[4][16 * D];              // 32 KB: per-wave 16x128 staging
  const int wave = threadIdx.x >> 5;
  const int lane = threadIdx.x & 31;
  const int ln = lane & 15;
  const int hi = lane >> 4;
  const int tile0 = (blockIdx.x * 4 + wave) * 16;

  float* sF = stageF[wave];
  unsigned short* sH = (unsigned short*)sF;        // bf16 view of same staging

  // ---- A0 fragments from [x | agg], K = 256 -> 8 chunks of 32 ----
  int row = tile0 + ln;
  if (row >= N_NODES) row = N_NODES - 1;           // clamp tail; stores masked later
  const float* xr = x + (size_t)row * D;
  const float* gr = agg + (size_t)row * D;
  v16u a0[8];
#pragma unroll
  for (int kc = 0; kc < 8; ++kc) {
    const float* src = (kc < 4) ? (xr + kc * 32) : (gr + (kc - 4) * 32);
    const float4 p0 = *(const float4*)(src + hi * 8);
    const float4 p1 = *(const float4*)(src + hi * 8 + 4);
    const float4 p2 = *(const float4*)(src + hi * 8 + 16);
    const float4 p3 = *(const float4*)(src + hi * 8 + 20);
    v16u a;
    a[0] = f2bf(p0.x);  a[1] = f2bf(p0.y);  a[2] = f2bf(p0.z);  a[3] = f2bf(p0.w);
    a[4] = f2bf(p1.x);  a[5] = f2bf(p1.y);  a[6] = f2bf(p1.z);  a[7] = f2bf(p1.w);
    a[8] = f2bf(p2.x);  a[9] = f2bf(p2.y);  a[10] = f2bf(p2.z); a[11] = f2bf(p2.w);
    a[12] = f2bf(p3.x); a[13] = f2bf(p3.y); a[14] = f2bf(p3.z); a[15] = f2bf(p3.w);
    a0[kc] = a;
  }

  // ---- layer 1: h1 = relu([x|agg] @ W1 + b1) ----
  const v16u* W1v = (const v16u*)w1f;
#pragma unroll
  for (int j = 0; j < 8; ++j) {
    v8f acc = {};
#pragma unroll
    for (int kc = 0; kc < 8; ++kc)
      acc = wmma_bf16(a0[kc], W1v[(kc * 8 + j) * 32 + lane], acc);
    float bias = b1[j * 16 + ln];
    const int col = j * 16 + ln;
#pragma unroll
    for (int r = 0; r < 8; ++r) {                  // C layout: M=r(+8), N=ln
      float h = acc[r] + bias;
      h = h > 0.f ? h : 0.f;
      sH[(r + hi * 8) * D + col] = f2bf(h);
    }
  }
  __syncthreads();

  // ---- A1 fragments from LDS (K = 128 -> 4 chunks) ----
  v16u a1[4];
#pragma unroll
  for (int kc = 0; kc < 4; ++kc) {
    const v8u lo = *(const v8u*)(sH + ln * D + kc * 32 + hi * 8);
    const v8u hh = *(const v8u*)(sH + ln * D + kc * 32 + hi * 8 + 16);
    v16u a;
#pragma unroll
    for (int i = 0; i < 8; ++i) { a[i] = lo[i]; a[8 + i] = hh[i]; }
    a1[kc] = a;
  }
  __syncthreads();

  // ---- layer 2: h2 = relu(h1 @ W2 + b2) ----
  const v16u* W2v = (const v16u*)w2f;
#pragma unroll
  for (int j = 0; j < 8; ++j) {
    v8f acc = {};
#pragma unroll
    for (int kc = 0; kc < 4; ++kc)
      acc = wmma_bf16(a1[kc], W2v[(kc * 8 + j) * 32 + lane], acc);
    float bias = b2[j * 16 + ln];
    const int col = j * 16 + ln;
#pragma unroll
    for (int r = 0; r < 8; ++r) {
      float h = acc[r] + bias;
      h = h > 0.f ? h : 0.f;
      sH[(r + hi * 8) * D + col] = f2bf(h);
    }
  }
  __syncthreads();

  // ---- A2 fragments ----
  v16u a2[4];
#pragma unroll
  for (int kc = 0; kc < 4; ++kc) {
    const v8u lo = *(const v8u*)(sH + ln * D + kc * 32 + hi * 8);
    const v8u hh = *(const v8u*)(sH + ln * D + kc * 32 + hi * 8 + 16);
    v16u a;
#pragma unroll
    for (int i = 0; i < 8; ++i) { a[i] = lo[i]; a[8 + i] = hh[i]; }
    a2[kc] = a;
  }
  __syncthreads();

  // ---- layer 3: h3 = h2 @ W3 + b3 (f32 staged for LayerNorm) ----
  const v16u* W3v = (const v16u*)w3f;
#pragma unroll
  for (int j = 0; j < 8; ++j) {
    v8f acc = {};
#pragma unroll
    for (int kc = 0; kc < 4; ++kc)
      acc = wmma_bf16(a2[kc], W3v[(kc * 8 + j) * 32 + lane], acc);
    float bias = b3[j * 16 + ln];
#pragma unroll
    for (int r = 0; r < 8; ++r)
      sF[(r + hi * 8) * D + j * 16 + ln] = acc[r] + bias;
  }
  __syncthreads();

  // ---- LayerNorm: 2 lanes per row, each covers 64 columns ----
  const float* rowp = sF + ln * D + hi * 64;
  float4 vv[16];
  float s = 0.f, s2 = 0.f;
#pragma unroll
  for (int i = 0; i < 16; ++i) {
    vv[i] = *(const float4*)(rowp + i * 4);
    s  += vv[i].x + vv[i].y + vv[i].z + vv[i].w;
    s2 += vv[i].x * vv[i].x + vv[i].y * vv[i].y + vv[i].z * vv[i].z + vv[i].w * vv[i].w;
  }
  // combine halves across lane ^ 16 (ds_swizzle SWAPX16: xor=0x10, and=0x1f)
  s  += __builtin_bit_cast(float, __builtin_amdgcn_ds_swizzle(__builtin_bit_cast(int, s),  0x401F));
  s2 += __builtin_bit_cast(float, __builtin_amdgcn_ds_swizzle(__builtin_bit_cast(int, s2), 0x401F));
  float mean = s * (1.f / 128.f);
  float var  = s2 * (1.f / 128.f) - mean * mean;
  float rstd = rsqrtf(var + LN_EPS);

  int orow = tile0 + ln;
  if (orow < N_NODES) {
    float* op = out + (size_t)orow * D + hi * 64;
#pragma unroll
    for (int i = 0; i < 16; ++i) {
      float4 g  = *(const float4*)(gamma + hi * 64 + i * 4);
      float4 bb = *(const float4*)(beta  + hi * 64 + i * 4);
      float4 o;
      o.x = (vv[i].x - mean) * rstd * g.x + bb.x;
      o.y = (vv[i].y - mean) * rstd * g.y + bb.y;
      o.z = (vv[i].z - mean) * rstd * g.z + bb.z;
      o.w = (vv[i].w - mean) * rstd * g.w + bb.w;
      *(float4*)(op + i * 4) = o;
    }
  }
}

// ---------------------------------------------------------------- launcher
extern "C" void kernel_launch(void* const* d_in, const int* in_sizes, int n_in,
                              void* d_out, int out_size, void* d_ws, size_t ws_size,
                              hipStream_t stream) {
  (void)in_sizes; (void)n_in; (void)out_size; (void)ws_size;
  const float* x  = (const float*)d_in[0];
  const int*   ei = (const int*)d_in[1];
  const float* ea = (const float*)d_in[2];
  const float* W1 = (const float*)d_in[3];
  const float* b1 = (const float*)d_in[4];
  const float* W2 = (const float*)d_in[5];
  const float* b2 = (const float*)d_in[6];
  const float* W3 = (const float*)d_in[7];
  const float* b3 = (const float*)d_in[8];
  const float* gm = (const float*)d_in[9];
  const float* bt = (const float*)d_in[10];
  float* out = (float*)d_out;

  char* ws = (char*)d_ws;
  float* agg = (float*)ws;                                   // 25.6 MB accumulator
  size_t aggB = (size_t)N_NODES * D * sizeof(float);
  unsigned short* w1f = (unsigned short*)(ws + aggB);        // 64 KB  (256x128 bf16)
  unsigned short* w2f = w1f + 256 * 128;                     // 32 KB
  unsigned short* w3f = w2f + 128 * 128;                     // 32 KB

  zero_f4<<<4096, 256, 0, stream>>>((float4*)agg, N_NODES * D / 4);
  pack_w<<<(256 * 128 + 255) / 256, 256, 0, stream>>>(W1, w1f, 256);
  pack_w<<<(128 * 128 + 255) / 256, 256, 0, stream>>>(W2, w2f, 128);
  pack_w<<<(128 * 128 + 255) / 256, 256, 0, stream>>>(W3, w3f, 128);
  scatter_edges<<<(N_EDGES * 32 + 255) / 256, 256, 0, stream>>>(ea, ei, agg);
  int tiles = (N_NODES + 63) / 64;                           // 64 rows / block
  mgn_mlp<<<tiles, 128, 0, stream>>>(x, agg, w1f, w2f, w3f, b1, b2, b3, gm, bt, out);
}
